// Gate_3032246911034
// MI455X (gfx1250) — matrix operand order repaired and checked by reading, not compiled, not run on hardware
//
#include <hip/hip_runtime.h>

typedef __attribute__((ext_vector_type(16))) __bf16 v16bf;
typedef __attribute__((ext_vector_type(8)))  __bf16 v8bf;
typedef __attribute__((ext_vector_type(8)))  float  v8f;

#define H    4096
#define E    256
#define TOPK 8
#define BK   256   // K-chunk staged in LDS
#define TM   16    // tokens per workgroup

__device__ __forceinline__ __bf16 f32_to_bf16(float f) {
    // round-to-nearest-even fp32 -> bf16
    unsigned u = __builtin_bit_cast(unsigned, f);
    u += 0x7FFFu + ((u >> 16) & 1u);
    unsigned short h = (unsigned short)(u >> 16);
    return __builtin_bit_cast(__bf16, h);
}

// ---------------------------------------------------------------------------
// Kernel 1: convert gate weight fp32 -> bf16 (2 MB into d_ws). 4 MB read,
// trivially memory bound; amortized across the whole gate GEMM.
// ---------------------------------------------------------------------------
__global__ void __launch_bounds__(256)
wcvt_kernel(const float* __restrict__ w, __bf16* __restrict__ o) {
    const int i = (blockIdx.x * 256 + threadIdx.x) * 4;
    const float4 f = *(const float4*)(w + i);
    o[i + 0] = f32_to_bf16(f.x);
    o[i + 1] = f32_to_bf16(f.y);
    o[i + 2] = f32_to_bf16(f.z);
    o[i + 3] = f32_to_bf16(f.w);
}

// ---------------------------------------------------------------------------
// Kernel 2: fused gate GEMM (bf16 WMMA) + sigmoid + bias + top-8 + normalize.
// Grid: T/16 workgroups. Block: 512 threads = 16 waves (wave32).
// Wave w computes the 16(token) x 16(expert) tile for experts [16w, 16w+16).
// ---------------------------------------------------------------------------
__global__ void __launch_bounds__(512)
moe_gate_kernel(const float* __restrict__ x, const __bf16* __restrict__ wbf,
                const float* __restrict__ bias, int* __restrict__ out_idx,
                float* __restrict__ out_w) {
    __shared__ __bf16 sA[TM * BK];       // 8 KB  : staged x tile (bf16)
    __shared__ float  sScore[TM * E];    // 16 KB : sigmoid(logit)+bias
    __shared__ float  sKw[TM * TOPK];
    __shared__ int    sKi[TM * TOPK];

    const int tid  = threadIdx.x;
    const int lane = tid & 31;
    const int wv   = tid >> 5;           // wave id 0..15
    const int t0   = blockIdx.x * TM;

    const int nsub = lane & 15;          // tile column (expert offset / token row)
    const int hi16 = lane >> 4;          // upper half-wave flag
    const int e0   = wv * 16;            // first expert of this wave's tile

    // Per-lane K offsets within a 32-wide WMMA k-step (ISA 7.12.2 layouts)
    const int aoff = hi16 ? 8  : 0;      // A: lanes 16-31 hold K=8..15 / 24..31
    const int boff = hi16 ? 8  : 0;      // B: lanes 16-31 hold K=8..15 / 24..31

    const float*  xrow = x   + (size_t)(t0 + wv) * H;       // wave wv stages token row wv
    const __bf16* brow = wbf + (size_t)(e0 + nsub) * H;     // this lane's expert row

    v8f acc = {};

    for (int kc = 0; kc < H; kc += BK) {
        __syncthreads();
        // Stage A: wave wv loads 8 consecutive floats per lane (coalesced 32B),
        // converts to bf16, stores 16B to LDS.
        {
            const float4 f0 = *(const float4*)(xrow + kc + lane * 8);
            const float4 f1 = *(const float4*)(xrow + kc + lane * 8 + 4);
            v8bf a;
            a[0] = f32_to_bf16(f0.x); a[1] = f32_to_bf16(f0.y);
            a[2] = f32_to_bf16(f0.z); a[3] = f32_to_bf16(f0.w);
            a[4] = f32_to_bf16(f1.x); a[5] = f32_to_bf16(f1.y);
            a[6] = f32_to_bf16(f1.z); a[7] = f32_to_bf16(f1.w);
            *(v8bf*)(&sA[wv * BK + lane * 8]) = a;
        }
        __syncthreads();

#pragma unroll
        for (int ks = 0; ks < BK; ks += 32) {
            // A fragment (16x32 bf16) from LDS: two ds_load_b128 per lane
            const v8bf alo = *(const v8bf*)(&sA[nsub * BK + ks + aoff]);
            const v8bf ahi = *(const v8bf*)(&sA[nsub * BK + ks + aoff + 16]);
            // B fragment (32x16 bf16) from L2-resident bf16 weights:
            // two runs of 8 bf16, 16 apart, per lane = two global_load_b128
            const v8bf blo = *(const v8bf*)(brow + kc + ks + boff);
            const v8bf bhi = *(const v8bf*)(brow + kc + ks + boff + 16);
            v16bf a, b;
#pragma unroll
            for (int i = 0; i < 8; ++i) {
                a[i] = alo[i]; a[8 + i] = ahi[i];
                b[i] = blo[i]; b[8 + i] = bhi[i];
            }
            acc = __builtin_amdgcn_wmma_f32_16x16x32_bf16(
                false, a, false, b, (short)0, acc, false, false);
        }
    }

    // Epilogue: sigmoid + bias correction, scatter tile into LDS score matrix.
    {
        const float bcorr = bias[e0 + nsub];
#pragma unroll
        for (int r = 0; r < 8; ++r) {
            const int m = r + hi16 * 8;                 // token row in tile
            const float sc = 1.0f / (1.0f + __expf(-acc[r])) + bcorr;
            sScore[m * E + e0 + nsub] = sc;
        }
    }
    __syncthreads();

    // Top-8 per token: wave wv handles token wv. Lane holds 8 of the 256
    // scores; 8 rounds of (local argmax over 8) + 5-step butterfly reduction.
    float v[8];
#pragma unroll
    for (int j = 0; j < 8; ++j) v[j] = sScore[wv * E + lane + 32 * j];

    float ksum = 0.0f;
#pragma unroll
    for (int k = 0; k < TOPK; ++k) {
        float bv = v[0];
        int   bi = lane;
#pragma unroll
        for (int j = 1; j < 8; ++j) {
            const int idx = lane + 32 * j;
            if (v[j] > bv) { bv = v[j]; bi = idx; }     // ascending idx => lowest wins ties
        }
#pragma unroll
        for (int off = 16; off > 0; off >>= 1) {
            const float ov = __shfl_xor(bv, off, 32);
            const int   oi = __shfl_xor(bi, off, 32);
            if (ov > bv || (ov == bv && oi < bi)) { bv = ov; bi = oi; }
        }
        // bv/bi now uniform across the wave
        ksum += bv;
        if (lane == 0) { sKw[wv * TOPK + k] = bv; sKi[wv * TOPK + k] = bi; }
        // knock out the winner (avoid dynamic register indexing)
        if ((bi & 31) == lane) {
            const int slot = bi >> 5;
#pragma unroll
            for (int j = 0; j < 8; ++j)
                if (j == slot) v[j] = -1e30f;
        }
    }
    __syncthreads();

    const float rscale = 2.5f / (ksum + 1e-20f);
    if (lane < TOPK) {
        const int t = t0 + wv;
        out_idx[t * TOPK + lane] = sKi[wv * TOPK + lane];
        out_w[t * TOPK + lane]   = sKw[wv * TOPK + lane] * rscale;
    }
}

// ---------------------------------------------------------------------------
extern "C" void kernel_launch(void* const* d_in, const int* in_sizes, int n_in,
                              void* d_out, int out_size, void* d_ws, size_t ws_size,
                              hipStream_t stream) {
    const float* x    = (const float*)d_in[0];
    const float* w    = (const float*)d_in[1];
    const float* bias = (const float*)d_in[2];
    const int T = in_sizes[0] / H;                     // 16384 tokens

    __bf16* wbf = (__bf16*)d_ws;                       // 2 MB bf16 weight in scratch

    int*   out_idx = (int*)d_out;                      // first T*8 int32 indices
    float* out_w   = (float*)d_out + (size_t)T * TOPK; // then T*8 f32 weights

    wcvt_kernel<<<(E * H) / (256 * 4), 256, 0, stream>>>(w, wbf);
    moe_gate_kernel<<<T / TM, 512, 0, stream>>>(x, wbf, bias, out_idx, out_w);
}
